// modConv_83073257439737
// MI455X (gfx1250) — matrix operand order repaired and checked by reading, not compiled, not run on hardware
//
#include <hip/hip_runtime.h>

typedef __attribute__((ext_vector_type(16))) _Float16 v16h;
typedef __attribute__((ext_vector_type(8)))  _Float16 v8h;
typedef __attribute__((ext_vector_type(8)))  float    v8f;

#define GAS __attribute__((address_space(1)))
typedef const GAS v8h* gv8h_p;

union V16 { v16h v; v8h h[2]; };

// ---------------------------------------------------------------------------
// Kernel 1: x NCHW fp32 (8,512,64,64) -> NHWC fp16 (8,64,64,512)
// ---------------------------------------------------------------------------
__global__ void xcvt_kernel(const float* __restrict__ x, _Float16* __restrict__ x16) {
  const int by = blockIdx.x;          // 512 blocks = 8 batches * 64 rows
  const int bb = by >> 6;
  const int yy = by & 63;
  __shared__ float tile[4][65];
  const int tid = threadIdx.x;
  for (int c0 = 0; c0 < 512; c0 += 4) {
    const int ci = tid >> 6;
    const int xx = tid & 63;
    tile[ci][xx] = x[(((size_t)bb * 512 + c0 + ci) * 64 + yy) * 64 + xx];
    __syncthreads();
    const int xo = tid >> 2;
    const int co = tid & 3;
    x16[(((size_t)bb * 64 + yy) * 64 + xo) * 512 + c0 + co] = (_Float16)tile[co][xo];
    __syncthreads();
  }
}

// ---------------------------------------------------------------------------
// Kernel 2: demodulate + pack per-class fp16 A; block 0 also clears zero page.
// ---------------------------------------------------------------------------
__global__ void wprep_kernel(const float* __restrict__ w, _Float16* __restrict__ Apack,
                             uint4* __restrict__ zpage) {
  const int o   = blockIdx.x;
  const int tid = threadIdx.x;
  if (o == 0) zpage[tid] = uint4{0u, 0u, 0u, 0u};   // 4 KB zero page

  __shared__ float red[256];
  __shared__ float coef_sh;
  const float* wo = w + (size_t)o * (512 * 9);

  float s = 0.f;
  for (int idx = tid; idx < 512 * 9; idx += 256) { float v = wo[idx]; s += v * v; }
  red[tid] = s;
  __syncthreads();
  for (int off = 128; off > 0; off >>= 1) {
    if (tid < off) red[tid] += red[tid + off];
    __syncthreads();
  }
  if (tid == 0) {
    const float ws2 = 1.0f / (512.f * 9.f);
    coef_sh = rsqrtf(red[0] * ws2 + 1e-6f) * sqrtf(ws2);
  }
  __syncthreads();
  const float coef = coef_sh;

  const size_t aoff[4] = {0, 524288, 786432, 1048576};
  const int    Kc[4]   = {2048, 1024, 1024, 512};
  for (int cls = 0; cls < 4; ++cls) {
    const int a = cls >> 1, b = cls & 1;
    const int nty = a ? 1 : 2, ntx = b ? 1 : 2;
    _Float16* A = Apack + aoff[cls] + (size_t)o * Kc[cls];
    for (int t = 0; t < nty * ntx; ++t) {
      const int ty = t / ntx, tx = t - (t / ntx) * ntx;
      const int wy = a ? 1 : (ty == 0 ? 2 : 0);
      const int wx = b ? 1 : (tx == 0 ? 2 : 0);
      for (int jj = 0; jj < 2; ++jj) {
        const int i = tid + 256 * jj;
        A[t * 512 + i] = (_Float16)(wo[(size_t)i * 9 + wy * 3 + wx] * coef);
      }
    }
  }
}

// ---------------------------------------------------------------------------
// Fragment load helpers (global address space, b128 pairs)
// ---------------------------------------------------------------------------
__device__ __forceinline__ void load_a4(V16 a[4], const _Float16* const aptr[4], int kofs) {
  #pragma unroll
  for (int m = 0; m < 4; ++m) {
    const _Float16* p = aptr[m] + kofs;
    a[m].h[0] = *(gv8h_p)(p);
    a[m].h[1] = *(gv8h_p)(p + 16);
  }
}
__device__ __forceinline__ void load_b2(V16 bfr[2], const _Float16* p0, const _Float16* p1, int kofs) {
  bfr[0].h[0] = *(gv8h_p)(p0 + kofs);
  bfr[0].h[1] = *(gv8h_p)(p0 + kofs + 16);
  bfr[1].h[0] = *(gv8h_p)(p1 + kofs);
  bfr[1].h[1] = *(gv8h_p)(p1 + kofs + 16);
}

// ---------------------------------------------------------------------------
// Kernel 3: implicit GEMM for one parity class.
// WG tile: M=128 x N=128.  8 waves in 2x4 grid; each wave M=64 x N=32
// (8 WMMA C tiles), double-buffered K pipeline, 32-K steps.
// ---------------------------------------------------------------------------
__global__ __launch_bounds__(256)
void gemm_up_kernel(const _Float16* __restrict__ x16,
                    const _Float16* __restrict__ Ac,
                    const _Float16* __restrict__ zpage,
                    float* __restrict__ y,
                    int a, int b, int Kc) {
  const int nqy = a ? 64 : 65;
  const int nqx = b ? 64 : 65;
  const int Npix = nqy * nqx;
  const int ntx = b ? 1 : 2;
  const int ntaps = (a ? 1 : 2) * ntx;

  const int tid    = threadIdx.x;
  const int lane   = tid & 31;
  const int wave   = tid >> 5;
  const int lane16 = lane & 15;
  const int laneHi = lane >> 4;
  const int waveM  = wave >> 2;       // 0..1
  const int waveN  = wave & 3;        // 0..3
  const int batch  = blockIdx.z;
  const int Mwave  = blockIdx.y * 128 + waveM * 64;

  int  nIdx[2], qy[2], qx[2];
  bool nval[2];
  #pragma unroll
  for (int j = 0; j < 2; ++j) {
    nIdx[j] = blockIdx.x * 128 + waveN * 32 + j * 16 + lane16;
    nval[j] = nIdx[j] < Npix;
    qy[j]   = nIdx[j] / nqx;
    qx[j]   = nIdx[j] - qy[j] * nqx;
  }

  const _Float16* aptr[4];
  #pragma unroll
  for (int m = 0; m < 4; ++m)
    aptr[m] = Ac + (size_t)(Mwave + 16 * m + lane16) * Kc + laneHi * 8;

  // Compute B base pointer for tap t, pixel sub-tile j (zero page if OOB).
  auto bbase = [&](int t, int j) -> const _Float16* {
    const int ty = t / ntx, tx = t - (t / ntx) * ntx;
    const int oy = a ? 0 : (ty == 0 ? -1 : 0);
    const int ox = b ? 0 : (tx == 0 ? -1 : 0);
    const int sy = qy[j] + oy, sx = qx[j] + ox;
    const bool v = nval[j] && ((unsigned)sy < 64u) && ((unsigned)sx < 64u);
    return v ? (x16 + ((size_t)((batch * 64 + sy) * 64 + sx) * 512 + laneHi * 8))
             : (zpage + laneHi * 8);
  };

  v8f acc[4][2] = {};

  const _Float16* bp[2] = { bbase(0, 0), bbase(0, 1) };
  for (int t = 0; t < ntaps; ++t) {
    // Prefetch next tap's activation rows into cache (global_prefetch_b8).
    const _Float16* bpn[2] = { bp[0], bp[1] };
    if (t + 1 < ntaps) {
      bpn[0] = bbase(t + 1, 0);
      bpn[1] = bbase(t + 1, 1);
      __builtin_prefetch(bpn[0], 0, 1);
      __builtin_prefetch(bpn[1], 0, 1);
    }

    const int kg = t * 512;
    V16 A2[2][4], B2[2][2];
    load_a4(A2[0], aptr, kg);
    load_b2(B2[0], bp[0], bp[1], 0);

    #pragma unroll
    for (int u = 0; u < 16; ++u) {
      const int cur = u & 1, nxt = cur ^ 1;
      if (u < 15) {                        // prefetch next K step into other buffer
        load_a4(A2[nxt], aptr, kg + (u + 1) * 32);
        load_b2(B2[nxt], bp[0], bp[1], (u + 1) * 32);
      }
      #pragma unroll
      for (int m = 0; m < 4; ++m) {
        acc[m][0] = __builtin_amdgcn_wmma_f32_16x16x32_f16(
            false, A2[cur][m].v, false, B2[cur][0].v, (short)0, acc[m][0], false, false);
        acc[m][1] = __builtin_amdgcn_wmma_f32_16x16x32_f16(
            false, A2[cur][m].v, false, B2[cur][1].v, (short)0, acc[m][1], false, false);
      }
    }
    bp[0] = bpn[0];
    bp[1] = bpn[1];
  }

  #pragma unroll
  for (int j = 0; j < 2; ++j) {
    if (nval[j]) {
      const int typix = 2 * qy[j] + a;
      const int txpix = 2 * qx[j] + b;
      float* ybase = y + ((size_t)(batch * 256) * 129 + typix) * 129 + txpix;
      #pragma unroll
      for (int m = 0; m < 4; ++m) {
        #pragma unroll
        for (int r = 0; r < 8; ++r) {
          const int M = Mwave + 16 * m + laneHi * 8 + r;
          ybase[(size_t)M * (129 * 129)] = acc[m][j][r];
        }
      }
    }
  }
}

// ---------------------------------------------------------------------------
// Kernel 4: depthwise 4x4 FIR blur, y(8,256,129,129) -> out(8,256,128,128).
// Each thread produces 4 outputs along x from a 7-sample row buffer.
// ---------------------------------------------------------------------------
__global__ void fir_kernel(const float* __restrict__ y, float* __restrict__ out) {
  const unsigned t = blockIdx.x * 256u + threadIdx.x;   // 2^23 threads
  const int v4 = (t & 31) * 4;          // 0,4,...,124
  const int u  = (t >> 5) & 127;
  const int c  = (t >> 12) & 255;
  const int bb = t >> 20;
  const float F[4] = {0.25f, 0.75f, 0.75f, 0.25f};
  const float* ych = y + (size_t)(bb * 256 + c) * (129 * 129);

  float s0 = 0.f, s1 = 0.f, s2 = 0.f, s3 = 0.f;
  #pragma unroll
  for (int j = 0; j < 4; ++j) {
    const int yy = u + j - 1;
    if ((unsigned)yy < 129u) {
      const float* row = ych + (size_t)yy * 129;
      float rbuf[7];
      #pragma unroll
      for (int k = 0; k < 7; ++k) {
        const int xx = v4 + k - 1;
        rbuf[k] = ((unsigned)xx < 129u) ? row[xx] : 0.f;
      }
      const float fj = F[j];
      #pragma unroll
      for (int k = 0; k < 4; ++k) {
        const float fjk = fj * F[k];
        s0 += fjk * rbuf[k];
        s1 += fjk * rbuf[k + 1];
        s2 += fjk * rbuf[k + 2];
        s3 += fjk * rbuf[k + 3];
      }
    }
  }
  float4* op = (float4*)(out + ((((size_t)bb * 256 + c) * 128 + u) * 128 + v4));
  *op = make_float4(s0, s1, s2, s3);
}

// ---------------------------------------------------------------------------
extern "C" void kernel_launch(void* const* d_in, const int* in_sizes, int n_in,
                              void* d_out, int out_size, void* d_ws, size_t ws_size,
                              hipStream_t stream) {
  (void)in_sizes; (void)n_in; (void)out_size; (void)ws_size;
  const float* x = (const float*)d_in[0];   // (8,512,64,64)
  const float* w = (const float*)d_in[1];   // (1,256,512,3,3)
  float* out = (float*)d_out;               // (8,256,128,128)

  char* ws = (char*)d_ws;
  _Float16* zpage = (_Float16*)ws;                                   //      4,096 B
  _Float16* x16   = (_Float16*)(ws + 4096);                          // 33,554,432 B
  _Float16* Apack = (_Float16*)(ws + 4096 + 33554432);               //  2,359,296 B
  float*    yws   = (float*)(ws + 4096 + 33554432 + 2359296);        // 136,323,072 B

  xcvt_kernel<<<dim3(512), dim3(256), 0, stream>>>(x, x16);
  wprep_kernel<<<dim3(256), dim3(256), 0, stream>>>(w, Apack, (uint4*)zpage);

  const int aoff[4] = {0, 524288, 786432, 1048576};  // halves
  const int Kc[4]   = {2048, 1024, 1024, 512};
  for (int cls = 0; cls < 4; ++cls) {
    const int a = cls >> 1, b = cls & 1;
    const int npix = (a ? 64 : 65) * (b ? 64 : 65);
    const int ntiles = (npix + 127) / 128;
    gemm_up_kernel<<<dim3(ntiles, 2, 8), dim3(256), 0, stream>>>(
        x16, Apack + aoff[cls], zpage, yws, a, b, Kc[cls]);
  }

  fir_kernel<<<dim3(32768), dim3(256), 0, stream>>>(yws, out);
}
